// PaliGemmaWithExpertModel_71433896067422
// MI455X (gfx1250) — matrix operand order repaired and checked by reading, not compiled, not run on hardware
//
#include <hip/hip_runtime.h>
#include <hip/hip_bf16.h>

// ---------------------------------------------------------------------------
// Types / helpers
// ---------------------------------------------------------------------------
typedef __attribute__((ext_vector_type(16))) __bf16 v16bf;
typedef __attribute__((ext_vector_type(8)))  float  v8f;

union Frag16 { uint4 q[2]; unsigned int u[8]; v16bf v; };
union Acc8   { float f[8]; v8f v; };

static __device__ __forceinline__ unsigned short f2bf(float f) {
  unsigned int u = __float_as_uint(f);
  unsigned int r = u + 0x7FFFu + ((u >> 16) & 1u);   // round-to-nearest-even
  return (unsigned short)(r >> 16);
}
static __device__ __forceinline__ float bf2f(unsigned short h) {
  return __uint_as_float(((unsigned int)h) << 16);
}
static __device__ __forceinline__ unsigned int pk2bf(float lo, float hi) {
  return (unsigned int)f2bf(lo) | ((unsigned int)f2bf(hi) << 16);
}
static __device__ __forceinline__ float gelu_tanh(float x) {
  float x3 = x * x * x;
  return 0.5f * x * (1.0f + tanhf(0.7978845608028654f * (x + 0.044715f * x3)));
}
static __device__ __forceinline__ int rowmap(int r, int seg, int stride, int off) {
  return (r / seg) * stride + off + r % seg;
}

// ---------------------------------------------------------------------------
// Adaptive modulation: mod[b, j] = cond[b,:] @ W[:, j] + bias[j]
// ---------------------------------------------------------------------------
__global__ void k_ada(const float* __restrict__ cond, const float* __restrict__ Wm,
                      const float* __restrict__ bias, float* __restrict__ out,
                      int CD, int NO, int B) {
  int t = blockIdx.x * blockDim.x + threadIdx.x;
  if (t >= B * NO) return;
  int b = t / NO, j = t % NO;
  float s = bias[j];
  const float* c = cond + (size_t)b * CD;
  for (int i = 0; i < CD; ++i) s += c[i] * Wm[(size_t)i * NO + j];
  out[t] = s;
}

// ---------------------------------------------------------------------------
// RMSNorm (plain): out = bf16( x * rsqrt(mean(x^2)+eps) * (1+w) )
// ---------------------------------------------------------------------------
__global__ __launch_bounds__(256) void k_rms_plain(const float* __restrict__ X,
                                                   const float* __restrict__ w,
                                                   unsigned short* __restrict__ out, int D) {
  int row = blockIdx.x;
  const float* x = X + (size_t)row * D;
  __shared__ float red[256];
  float ss = 0.f;
  for (int i = threadIdx.x; i < D; i += 256) { float v = x[i]; ss += v * v; }
  red[threadIdx.x] = ss; __syncthreads();
  for (int s = 128; s > 0; s >>= 1) {
    if (threadIdx.x < s) red[threadIdx.x] += red[threadIdx.x + s];
    __syncthreads();
  }
  float rs = rsqrtf(red[0] / (float)D + 1e-6f);
  for (int i = threadIdx.x; i < D; i += 256)
    out[(size_t)row * D + i] = f2bf(x[i] * rs * (1.0f + w[i]));
}

// ---------------------------------------------------------------------------
// AdaRMS: out = bf16( rms(x) * (1+scale_b) + shift_b ), mod row layout [scale|shift|gate]
// ---------------------------------------------------------------------------
__global__ __launch_bounds__(256) void k_rms_ada(const float* __restrict__ X,
                                                 const float* __restrict__ mod, int mstride,
                                                 int rows_per_b,
                                                 unsigned short* __restrict__ out, int D) {
  int row = blockIdx.x;
  int b = row / rows_per_b;
  const float* x = X + (size_t)row * D;
  const float* mo = mod + (size_t)b * mstride;
  __shared__ float red[256];
  float ss = 0.f;
  for (int i = threadIdx.x; i < D; i += 256) { float v = x[i]; ss += v * v; }
  red[threadIdx.x] = ss; __syncthreads();
  for (int s = 128; s > 0; s >>= 1) {
    if (threadIdx.x < s) red[threadIdx.x] += red[threadIdx.x + s];
    __syncthreads();
  }
  float rs = rsqrtf(red[0] / (float)D + 1e-6f);
  for (int i = threadIdx.x; i < D; i += 256)
    out[(size_t)row * D + i] = f2bf(x[i] * rs * (1.0f + mo[i]) + mo[D + i]);
}

// ---------------------------------------------------------------------------
// Generic bf16 WMMA GEMM.  C[M,N] = A_bf16[M,K] @ W_f32->bf16[K,N]
// Block tile 128x128, 8 waves (4M x 2N), wave tile 32x64, K-step 32.
// Epilogue: optional gate (per b,col), optional residual (logical rows),
// fp32 and/or bf16 output with output-row remapping.
// ---------------------------------------------------------------------------
__global__ __launch_bounds__(256) void k_gemm(
    const unsigned short* __restrict__ A, int K,
    const float* __restrict__ W, int N,
    float* __restrict__ outF, unsigned short* __restrict__ outB, int ldo,
    const float* __restrict__ res,
    const float* __restrict__ gate, int gate_stride, int rows_per_b,
    int a_seg, int a_stride, int a_off,
    int c_seg, int c_stride, int c_off) {
  __shared__ unsigned short As[128][32];   // [m][k] bf16 packed
  __shared__ unsigned short Bs[128][32];   // [n][k] bf16 packed (transposed weight panel)

  const int tid  = threadIdx.x;
  const int lane = tid & 31, lr = lane & 15, hi = lane >> 4;
  const int wv = tid >> 5, wm = wv >> 1, wn = wv & 1;
  const int m0 = blockIdx.y * 128;
  const int n0 = blockIdx.x * 128;

  Acc8 acc[2][4];
#pragma unroll
  for (int s = 0; s < 2; ++s)
#pragma unroll
    for (int t = 0; t < 4; ++t)
#pragma unroll
      for (int r = 0; r < 8; ++r) acc[s][t].f[r] = 0.f;

  // A-staging: thread -> (row, 16-elem half)
  const int arow = tid >> 1, apart = tid & 1;
  const int gArow = rowmap(m0 + arow, a_seg, a_stride, a_off);
  const unsigned short* aSrc = A + (size_t)gArow * K + apart * 16;
  // B-staging: thread -> (n col, 16-k half); strided fp32 column reads -> bf16 pack
  const int bn = tid & 127, bh = tid >> 7;

  for (int kk = 0; kk < K; kk += 32) {
    __syncthreads();
    { // stage A (32 bytes per thread)
      const uint4* s = (const uint4*)(aSrc + kk);
      uint4* d = (uint4*)&As[arow][apart * 16];
      d[0] = s[0]; d[1] = s[1];
    }
    { // stage B transposed, fp32 -> bf16
      const float* wp = W + (size_t)(kk + bh * 16) * N + (n0 + bn);
      unsigned int pk[8];
#pragma unroll
      for (int j = 0; j < 8; ++j)
        pk[j] = pk2bf(wp[(size_t)(2 * j) * N], wp[(size_t)(2 * j + 1) * N]);
      uint4* d = (uint4*)&Bs[bn][bh * 16];
      d[0] = make_uint4(pk[0], pk[1], pk[2], pk[3]);
      d[1] = make_uint4(pk[4], pk[5], pk[6], pk[7]);
    }
    __syncthreads();

    Frag16 af[2], bf[4];
#pragma unroll
    for (int s = 0; s < 2; ++s) { // A layout: dwords v<4 -> K=2v+8hi ; v>=4 -> 16+2(v-4)+8hi
      const unsigned int* rp = (const unsigned int*)&As[wm * 32 + s * 16 + lr][0];
      af[s].q[0] = *(const uint4*)(rp + 4 * hi);
      af[s].q[1] = *(const uint4*)(rp + 8 + 4 * hi);
    }
#pragma unroll
    for (int t = 0; t < 4; ++t) { // B layout: dwords v<4 -> K=2v+8hi ; v>=4 -> 16+2(v-4)+8hi
      const unsigned int* rp = (const unsigned int*)&Bs[wn * 64 + t * 16 + lr][0];
      bf[t].q[0] = *(const uint4*)(rp + 4 * hi);
      bf[t].q[1] = *(const uint4*)(rp + 8 + 4 * hi);
    }
#pragma unroll
    for (int s = 0; s < 2; ++s)
#pragma unroll
      for (int t = 0; t < 4; ++t)
        acc[s][t].v = __builtin_amdgcn_wmma_f32_16x16x32_bf16(
            false, af[s].v, false, bf[t].v, (short)0, acc[s][t].v, false, false);
  }

  // Epilogue. C layout: vgpr r -> row r+8hi, col = lane&15.
#pragma unroll
  for (int s = 0; s < 2; ++s)
#pragma unroll
    for (int t = 0; t < 4; ++t) {
      int col = n0 + wn * 64 + t * 16 + lr;
#pragma unroll
      for (int r = 0; r < 8; ++r) {
        int rl = m0 + wm * 32 + s * 16 + r + 8 * hi;   // logical row
        float v = acc[s][t].f[r];
        if (gate) v *= gate[(size_t)(rl / rows_per_b) * gate_stride + col];
        if (res)  v += res[(size_t)rl * ldo + col];
        int orow = rowmap(rl, c_seg, c_stride, c_off);
        if (outF) outF[(size_t)orow * ldo + col] = v;
        if (outB) outB[(size_t)orow * ldo + col] = f2bf(v);
      }
    }
}

// ---------------------------------------------------------------------------
// Fused gate/up dual-GEMM:  out = bf16( gelu_tanh(A@Wg) * (A@Wu) )
// Block tile 128x64, 8 waves (4M x 2N), wave tile 32x32.
// ---------------------------------------------------------------------------
__global__ __launch_bounds__(256) void k_gemm_dual(
    const unsigned short* __restrict__ A, int K,
    const float* __restrict__ Wg, const float* __restrict__ Wu, int N,
    unsigned short* __restrict__ outB) {
  __shared__ unsigned short As[128][32];
  __shared__ unsigned short Bg[64][32];
  __shared__ unsigned short Bu[64][32];

  const int tid  = threadIdx.x;
  const int lane = tid & 31, lr = lane & 15, hi = lane >> 4;
  const int wv = tid >> 5, wm = wv >> 1, wn = wv & 1;
  const int m0 = blockIdx.y * 128;
  const int n0 = blockIdx.x * 64;

  Acc8 ag[2][2], au[2][2];
#pragma unroll
  for (int s = 0; s < 2; ++s)
#pragma unroll
    for (int t = 0; t < 2; ++t)
#pragma unroll
      for (int r = 0; r < 8; ++r) { ag[s][t].f[r] = 0.f; au[s][t].f[r] = 0.f; }

  const int arow = tid >> 1, apart = tid & 1;
  const unsigned short* aSrc = A + (size_t)(m0 + arow) * K + apart * 16;
  const int mat = tid & 1, bn = (tid >> 1) & 63, bh = tid >> 7;
  const float* wsel = mat ? Wu : Wg;
  unsigned short (*bdst)[32] = mat ? Bu : Bg;

  for (int kk = 0; kk < K; kk += 32) {
    __syncthreads();
    {
      const uint4* s = (const uint4*)(aSrc + kk);
      uint4* d = (uint4*)&As[arow][apart * 16];
      d[0] = s[0]; d[1] = s[1];
    }
    {
      const float* wp = wsel + (size_t)(kk + bh * 16) * N + (n0 + bn);
      unsigned int pk[8];
#pragma unroll
      for (int j = 0; j < 8; ++j)
        pk[j] = pk2bf(wp[(size_t)(2 * j) * N], wp[(size_t)(2 * j + 1) * N]);
      uint4* d = (uint4*)&bdst[bn][bh * 16];
      d[0] = make_uint4(pk[0], pk[1], pk[2], pk[3]);
      d[1] = make_uint4(pk[4], pk[5], pk[6], pk[7]);
    }
    __syncthreads();

    Frag16 af[2], gf[2], uf[2];
#pragma unroll
    for (int s = 0; s < 2; ++s) {
      const unsigned int* rp = (const unsigned int*)&As[wm * 32 + s * 16 + lr][0];
      af[s].q[0] = *(const uint4*)(rp + 4 * hi);
      af[s].q[1] = *(const uint4*)(rp + 8 + 4 * hi);
    }
#pragma unroll
    for (int t = 0; t < 2; ++t) {
      const unsigned int* rg = (const unsigned int*)&Bg[wn * 32 + t * 16 + lr][0];
      gf[t].q[0] = *(const uint4*)(rg + 4 * hi);
      gf[t].q[1] = *(const uint4*)(rg + 8 + 4 * hi);
      const unsigned int* ru = (const unsigned int*)&Bu[wn * 32 + t * 16 + lr][0];
      uf[t].q[0] = *(const uint4*)(ru + 4 * hi);
      uf[t].q[1] = *(const uint4*)(ru + 8 + 4 * hi);
    }
#pragma unroll
    for (int s = 0; s < 2; ++s)
#pragma unroll
      for (int t = 0; t < 2; ++t) {
        ag[s][t].v = __builtin_amdgcn_wmma_f32_16x16x32_bf16(
            false, af[s].v, false, gf[t].v, (short)0, ag[s][t].v, false, false);
        au[s][t].v = __builtin_amdgcn_wmma_f32_16x16x32_bf16(
            false, af[s].v, false, uf[t].v, (short)0, au[s][t].v, false, false);
      }
  }

#pragma unroll
  for (int s = 0; s < 2; ++s)
#pragma unroll
    for (int t = 0; t < 2; ++t) {
      int col = n0 + wn * 32 + t * 16 + lr;
#pragma unroll
      for (int r = 0; r < 8; ++r) {
        int row = m0 + wm * 32 + s * 16 + r + 8 * hi;
        outB[(size_t)row * N + col] = f2bf(gelu_tanh(ag[s][t].f[r]) * au[s][t].f[r]);
      }
    }
}

// ---------------------------------------------------------------------------
// RoPE in-place on bf16 buffer [B,S,Wd] (heads of 256, pairs (d, d+128))
// ---------------------------------------------------------------------------
__global__ void k_rope(unsigned short* __restrict__ buf, const int* __restrict__ pos_ids,
                       int S, int Wd, int total) {
  int t = blockIdx.x * blockDim.x + threadIdx.x;
  if (t >= total) return;
  int pairs = Wd >> 1;
  int row = t / pairs, pi = t % pairs;
  int h = pi >> 7, d = pi & 127;
  int b = row / S, s = row % S;
  float pos = (float)pos_ids[b * S + s];
  float inv = exp2f(-(float)d * (13.287712379549449f / 128.0f)); // 10000^(-d/128)
  float ang = pos * inv;
  float sn, cs;
  __sincosf(ang, &sn, &cs);
  unsigned short* p = buf + (size_t)row * Wd + h * 256 + d;
  float x1 = bf2f(p[0]), x2 = bf2f(p[128]);
  p[0]   = f2bf(x1 * cs - x2 * sn);
  p[128] = f2bf(x2 * cs + x1 * sn);
}

// ---------------------------------------------------------------------------
// Flash-style WMMA attention. One block = (b, h, 64-query tile), 4 waves x 16 q.
// Streams 32-key tiles; online softmax; PV via WMMA with V staged transposed.
// ---------------------------------------------------------------------------
__global__ __launch_bounds__(128) void k_attn(
    const unsigned short* __restrict__ Q, const unsigned short* __restrict__ Kb,
    const unsigned short* __restrict__ Vb, const float* __restrict__ mask,
    unsigned short* __restrict__ att, int S) {
  extern __shared__ unsigned short smem[];
  unsigned short* Qs = smem;                 // [64][256]
  unsigned short* Ks = Qs + 64 * 256;        // [32][256]
  unsigned short* Vt = Ks + 32 * 256;        // [256][32] (transposed V tile)
  unsigned short* Ps = Vt + 256 * 32;        // [4][16][32] per-wave P tile

  const int tid  = threadIdx.x;
  const int lane = tid & 31, lr = lane & 15, hi = lane >> 4;
  const int wv = tid >> 5;
  const int qblk = blockIdx.x, h = blockIdx.y, b = blockIdx.z;
  const int q0 = qblk * 64;

  // stage Q tile (64 x 256 bf16)
  for (int i = 0; i < 16; ++i) {
    int idx = tid + i * 128;
    int row = idx >> 5, c4 = idx & 31;
    *((uint4*)(Qs + row * 256) + c4) =
        *((const uint4*)(Q + (size_t)(b * S + q0 + row) * 2048 + h * 256) + c4);
  }

  Acc8 O[16];
  float m[8], l[8];
#pragma unroll
  for (int t = 0; t < 16; ++t)
#pragma unroll
    for (int r = 0; r < 8; ++r) O[t].f[r] = 0.f;
#pragma unroll
  for (int r = 0; r < 8; ++r) { m[r] = -1e30f; l[r] = 0.f; }

  const int nkt = S / 32;
  for (int kt = 0; kt < nkt; ++kt) {
    __syncthreads();
    for (int i = 0; i < 8; ++i) { // K tile (32 x 256)
      int idx = tid + i * 128;
      int row = idx >> 5, c4 = idx & 31;
      *((uint4*)(Ks + row * 256) + c4) =
          *((const uint4*)(Kb + (size_t)(b * S + kt * 32 + row) * 256) + c4);
    }
    for (int i = 0; i < 64; ++i) { // V tile transposed: Vt[d][key]
      int idx = tid * 64 + i;
      int key = idx >> 8, d = idx & 255;
      Vt[d * 32 + key] = Vb[(size_t)(b * S + kt * 32 + key) * 256 + d];
    }
    __syncthreads();

    // scores: Q(16x256) . K^T -> two 16x16 tiles (keys 0-15, 16-31)
    Acc8 sc[2];
    for (int nt = 0; nt < 2; ++nt) {
      Acc8 c;
#pragma unroll
      for (int r = 0; r < 8; ++r) c.f[r] = 0.f;
#pragma unroll
      for (int ks = 0; ks < 8; ++ks) {
        Frag16 a, bb;
        const unsigned int* ap = (const unsigned int*)(Qs + (wv * 16 + lr) * 256) + ks * 16;
        a.q[0] = *(const uint4*)(ap + 4 * hi);
        a.q[1] = *(const uint4*)(ap + 8 + 4 * hi);
        const unsigned int* bp = (const unsigned int*)(Ks + (nt * 16 + lr) * 256) + ks * 16;
        bb.q[0] = *(const uint4*)(bp + 4 * hi);
        bb.q[1] = *(const uint4*)(bp + 8 + 4 * hi);
        c.v = __builtin_amdgcn_wmma_f32_16x16x32_bf16(
            false, a.v, false, bb.v, (short)0, c.v, false, false);
      }
      sc[nt] = c;
    }
    // scale + additive mask
#pragma unroll
    for (int nt = 0; nt < 2; ++nt)
#pragma unroll
      for (int r = 0; r < 8; ++r) {
        int qrow = q0 + wv * 16 + r + 8 * hi;
        int kcol = kt * 32 + nt * 16 + lr;
        sc[nt].f[r] = sc[nt].f[r] * 0.0625f + mask[((size_t)b * S + qrow) * S + kcol];
      }
    // online softmax (row reduction within 16-lane halves)
#pragma unroll
    for (int r = 0; r < 8; ++r) {
      float x = fmaxf(sc[0].f[r], sc[1].f[r]);
      for (int off = 1; off < 16; off <<= 1) x = fmaxf(x, __shfl_xor(x, off, 32));
      float mn  = fmaxf(m[r], x);
      float fac = __expf(m[r] - mn);
      m[r] = mn;
      sc[0].f[r] = __expf(sc[0].f[r] - mn);
      sc[1].f[r] = __expf(sc[1].f[r] - mn);
      float rs = sc[0].f[r] + sc[1].f[r];
      for (int off = 1; off < 16; off <<= 1) rs += __shfl_xor(rs, off, 32);
      l[r] = l[r] * fac + rs;
#pragma unroll
      for (int t = 0; t < 16; ++t) O[t].f[r] *= fac;
    }
    // P (C layout) -> LDS -> A layout for PV
    unsigned short* pw = Ps + wv * (16 * 32);
#pragma unroll
    for (int nt = 0; nt < 2; ++nt)
#pragma unroll
      for (int r = 0; r < 8; ++r)
        pw[(r + 8 * hi) * 32 + nt * 16 + lr] = f2bf(sc[nt].f[r]);
    __syncthreads();

    Frag16 pa;
    const unsigned int* pp = (const unsigned int*)(pw + lr * 32);
    pa.q[0] = *(const uint4*)(pp + 4 * hi);
    pa.q[1] = *(const uint4*)(pp + 8 + 4 * hi);
#pragma unroll
    for (int t = 0; t < 16; ++t) {
      Frag16 vb2;
      const unsigned int* vp = (const unsigned int*)(Vt + (t * 16 + lr) * 32);
      vb2.q[0] = *(const uint4*)(vp + 4 * hi);
      vb2.q[1] = *(const uint4*)(vp + 8 + 4 * hi);
      O[t].v = __builtin_amdgcn_wmma_f32_16x16x32_bf16(
          false, pa.v, false, vb2.v, (short)0, O[t].v, false, false);
    }
  }

  // normalize + store att[b, q, h*256 + d] (bf16)
#pragma unroll
  for (int t = 0; t < 16; ++t)
#pragma unroll
    for (int r = 0; r < 8; ++r) {
      int qrow = q0 + wv * 16 + r + 8 * hi;
      int col  = t * 16 + lr;
      att[(size_t)(b * S + qrow) * 2048 + h * 256 + col] = f2bf(O[t].f[r] / l[r]);
    }
}

// ---------------------------------------------------------------------------
// Orchestration
// ---------------------------------------------------------------------------
extern "C" void kernel_launch(void* const* d_in, const int* in_sizes, int n_in,
                              void* d_out, int out_size, void* d_ws, size_t ws_size,
                              hipStream_t stream) {
  (void)in_sizes; (void)n_in; (void)out_size; (void)ws_size;
  const int S = 1088;

  const float* prefix  = (const float*)d_in[0];
  const float* suffix  = (const float*)d_in[1];
  const float* cond    = (const float*)d_in[2];
  const float* mask    = (const float*)d_in[3];
  const int*   pos     = (const int*)d_in[4];
  const float* v_ln1_w = (const float*)d_in[5];
  const float* v_wq    = (const float*)d_in[6];
  const float* v_wk    = (const float*)d_in[7];
  const float* v_wv    = (const float*)d_in[8];
  const float* v_wo    = (const float*)d_in[9];
  const float* v_ln2_w = (const float*)d_in[10];
  const float* v_wgate = (const float*)d_in[11];
  const float* v_wup   = (const float*)d_in[12];
  const float* v_wdown = (const float*)d_in[13];
  const float* e_ln1_W = (const float*)d_in[14];
  const float* e_ln1_b = (const float*)d_in[15];
  const float* e_wq    = (const float*)d_in[16];
  const float* e_wk    = (const float*)d_in[17];
  const float* e_wv    = (const float*)d_in[18];
  const float* e_wo    = (const float*)d_in[19];
  const float* e_ln2_W = (const float*)d_in[20];
  const float* e_ln2_b = (const float*)d_in[21];
  const float* e_wgate = (const float*)d_in[22];
  const float* e_wup   = (const float*)d_in[23];
  const float* e_wdown = (const float*)d_in[24];
  float* outP = (float*)d_out;

  // workspace carve-out (256B aligned)
  char* ws = (char*)d_ws;
  size_t off = 0;
  auto alloc = [&](size_t bytes) -> void* {
    void* p = ws + off;
    off = (off + bytes + 255) & ~(size_t)255;
    return p;
  };
  unsigned short* hv   = (unsigned short*)alloc((size_t)2048 * 2048 * 2); // rms(prefix)
  unsigned short* he   = (unsigned short*)alloc((size_t)128 * 1024 * 2);  // adarms(suffix)
  unsigned short* qb   = (unsigned short*)alloc((size_t)2 * S * 2048 * 2);
  unsigned short* kb   = (unsigned short*)alloc((size_t)2 * S * 256 * 2);
  unsigned short* vb   = (unsigned short*)alloc((size_t)2 * S * 256 * 2);
  unsigned short* attb = (unsigned short*)alloc((size_t)2 * S * 2048 * 2);
  float*          mod1 = (float*)alloc((size_t)2 * 3072 * 4);
  float*          mod2 = (float*)alloc((size_t)2 * 3072 * 4);
  float*          resv = (float*)alloc((size_t)2048 * 2048 * 4);
  float*          rese = (float*)alloc((size_t)128 * 1024 * 4);
  unsigned short* hv2  = (unsigned short*)alloc((size_t)2048 * 2048 * 2);
  unsigned short* he2  = (unsigned short*)alloc((size_t)128 * 1024 * 2);
  unsigned short* tv   = (unsigned short*)alloc((size_t)2048 * 16384 * 2);
  unsigned short* te   = (unsigned short*)alloc((size_t)128 * 4096 * 2);

  // 1) adaptive modulation vectors
  k_ada<<<24, 256, 0, stream>>>(cond, e_ln1_W, e_ln1_b, mod1, 1024, 3072, 2);
  k_ada<<<24, 256, 0, stream>>>(cond, e_ln2_W, e_ln2_b, mod2, 1024, 3072, 2);

  // 2) input norms -> bf16 activations
  k_rms_plain<<<2048, 256, 0, stream>>>(prefix, v_ln1_w, hv, 2048);
  k_rms_ada<<<128, 256, 0, stream>>>(suffix, mod1, 3072, 64, he, 1024);

  // 3) QKV projections (output rows remapped into [B, S=1088, *] concat layout)
  k_gemm<<<dim3(16, 16), 256, 0, stream>>>(hv, 2048, v_wq, 2048, nullptr, qb, 2048,
      nullptr, nullptr, 1, 1, 2048, 2048, 0, 1024, 1088, 0);
  k_gemm<<<dim3(16, 1), 256, 0, stream>>>(he, 1024, e_wq, 2048, nullptr, qb, 2048,
      nullptr, nullptr, 1, 1, 128, 128, 0, 64, 1088, 1024);
  k_gemm<<<dim3(2, 16), 256, 0, stream>>>(hv, 2048, v_wk, 256, nullptr, kb, 256,
      nullptr, nullptr, 1, 1, 2048, 2048, 0, 1024, 1088, 0);
  k_gemm<<<dim3(2, 1), 256, 0, stream>>>(he, 1024, e_wk, 256, nullptr, kb, 256,
      nullptr, nullptr, 1, 1, 128, 128, 0, 64, 1088, 1024);
  k_gemm<<<dim3(2, 16), 256, 0, stream>>>(hv, 2048, v_wv, 256, nullptr, vb, 256,
      nullptr, nullptr, 1, 1, 2048, 2048, 0, 1024, 1088, 0);
  k_gemm<<<dim3(2, 1), 256, 0, stream>>>(he, 1024, e_wv, 256, nullptr, vb, 256,
      nullptr, nullptr, 1, 1, 128, 128, 0, 64, 1088, 1024);

  // 4) RoPE on q and k
  k_rope<<<8704, 256, 0, stream>>>(qb, pos, S, 2048, 2 * S * 1024);
  k_rope<<<1088, 256, 0, stream>>>(kb, pos, S, 256, 2 * S * 128);

  // 5) attention (dynamic LDS: Qs+Ks+Vt+Ps = 69632 B)
  k_attn<<<dim3(17, 8, 2), 128, 69632, stream>>>(qb, kb, vb, mask, attb, S);

  // 6) output projections + residuals (expert path gated by mod1[:, 2D:3D])
  k_gemm<<<dim3(16, 16), 256, 0, stream>>>(attb, 2048, v_wo, 2048, resv, nullptr, 2048,
      prefix, nullptr, 1, 1, 1024, 1088, 0, 2048, 2048, 0);
  k_gemm<<<dim3(8, 1), 256, 0, stream>>>(attb, 2048, e_wo, 1024, rese, nullptr, 1024,
      suffix, mod1 + 2048, 3072, 64, 64, 1088, 1024, 128, 128, 0);

  // 7) post-attention norms
  k_rms_plain<<<2048, 256, 0, stream>>>(resv, v_ln2_w, hv2, 2048);
  k_rms_ada<<<128, 256, 0, stream>>>(rese, mod2, 3072, 64, he2, 1024);

  // 8) fused gate/up MLP GEMMs
  k_gemm_dual<<<dim3(256, 16), 256, 0, stream>>>(hv2, 2048, v_wgate, v_wup, 16384, tv);
  k_gemm_dual<<<dim3(64, 1), 256, 0, stream>>>(he2, 1024, e_wgate, e_wup, 4096, te);

  // 9) down projections + residuals -> final fp32 outputs
  k_gemm<<<dim3(16, 16), 256, 0, stream>>>(tv, 16384, v_wdown, 2048, outP, nullptr, 2048,
      resv, nullptr, 1, 1, 2048, 2048, 0, 2048, 2048, 0);
  k_gemm<<<dim3(8, 1), 256, 0, stream>>>(te, 4096, e_wdown, 1024, outP + 4194304, nullptr, 1024,
      rese, mod2 + 2048, 3072, 64, 128, 128, 0, 128, 128, 0);
}
